// DecoderRNN_13615046329063
// MI455X (gfx1250) — compile-verified, matrix operand order checked by reading, not run on hardware
//
#include <hip/hip_runtime.h>
#include <hip/hip_bf16.h>
#include <math.h>

typedef __attribute__((ext_vector_type(16))) _Float16 v16h;
typedef __attribute__((ext_vector_type(8)))  _Float16 v8h;
typedef __attribute__((ext_vector_type(8)))  float    v8f;
typedef __attribute__((ext_vector_type(4)))  unsigned int u32x4;
typedef __attribute__((ext_vector_type(8)))  int      i32x8;
typedef __attribute__((ext_vector_type(4)))  int      i32x4;

#define H_DIM   1024
#define C_CLS   151
#define E_DIM   200
#define E_PAD   224      // 200 padded to 7*32
#define DIN     2048
#define KTI     2272     // DIN + E_PAD (WiT row stride in halves)
#define T_STEPS 32
#define B_ROWS  128
#define N6H     6144
#define N5H     5120
#define NWO     160      // 151 padded to 16*10
#define LDS_PITCH 40     // 32 + 8 halves: lane stride = 20 banks -> conflict-free
#define TILE_HALVES (6 * 16 * LDS_PITCH)   // 3840 halves = 7680 B per buffer

#if __has_builtin(__builtin_amdgcn_tensor_load_to_lds) && \
    __has_builtin(__builtin_amdgcn_s_wait_tensorcnt)
#define USE_TDM 1
#else
#define USE_TDM 0
#endif

// ---------------- WMMA helpers ----------------

__device__ __forceinline__ v8f zero8() {
  v8f z;
#pragma unroll
  for (int i = 0; i < 8; ++i) z[i] = 0.0f;
  return z;
}

// Load one 16x32 f16 fragment in the CDNA5 WMMA A/B register layout from
// row-major global memory (ld in halves). Lanes 0-15: row r0+lane, K groups
// {0..7,16..23}; lanes 16-31: same rows, K groups {8..15,24..31}.
__device__ __forceinline__ v16h load_frag16(const _Float16* __restrict__ p,
                                            int ld, int r0, int k0) {
  const int lane = threadIdx.x & 31;
  const _Float16* base =
      p + (size_t)(r0 + (lane & 15)) * (size_t)ld + k0 + ((lane >> 4) << 3);
  v8h lo = *(const v8h*)(base);
  v8h hi = *(const v8h*)(base + 16);
  v16h out;
#pragma unroll
  for (int i = 0; i < 8; ++i) { out[i] = lo[i]; out[i + 8] = hi[i]; }
  return out;
}

// Same fragment layout but reading a staged 16xLDS_PITCH tile from LDS.
__device__ __forceinline__ v16h lds_frag16(const _Float16* t) {
  const int lane = threadIdx.x & 31;
  const _Float16* base = t + (lane & 15) * LDS_PITCH + ((lane >> 4) << 3);
  v8h lo = *(const v8h*)(base);
  v8h hi = *(const v8h*)(base + 16);
  v16h out;
#pragma unroll
  for (int i = 0; i < 8; ++i) { out[i] = lo[i]; out[i + 8] = hi[i]; }
  return out;
}

__device__ __forceinline__ v8f wmma_f16(v16h a, v16h b, v8f c) {
  return __builtin_amdgcn_wmma_f32_16x16x32_f16(
      false, a, false, b, (short)0, c, false, false);
}

__device__ __forceinline__ float sigmoidf_(float x) {
  return 1.0f / (1.0f + __expf(-x));
}

#if USE_TDM
// Issue one TDM 3-D tile load: tile_dim0 = 32 halves (K, contiguous),
// tile_dim1 = 16 rows (stride rowStride halves), tile_dim2 = groups gate
// segments (stride grpStride halves). LDS destination gets 4 DWORDs of pad
// after every 16 data DWORDs (pad_interval=3, pad_amount=3) -> pitch 40
// halves, matching lds_frag16's conflict-free layout.
__device__ __forceinline__ void tdm_load_tile(unsigned ldsAddr,
                                              const _Float16* gptr,
                                              int groups,
                                              unsigned rowStride,
                                              unsigned grpStride) {
  unsigned long long ga = (unsigned long long)gptr;
  u32x4 g0;
  g0[0] = 1u;                                   // count=1 (valid user D#)
  g0[1] = ldsAddr;                              // lds_addr (bytes)
  g0[2] = (unsigned)(ga & 0xffffffffu);         // global_addr[31:0]
  g0[3] = (unsigned)((ga >> 32) & 0x01ffffffu)  // global_addr[56:32]
          | (2u << 30);                         // type = 2 ("image")
  const unsigned td0 = 0x40000000u, td1 = 0x40000000u;  // huge: no OOB clip
  i32x8 g1;
  g1[0] = (int)((1u << 16)      // data_size = 1 -> 2 bytes
              | (1u << 20)      // pad_enable
              | (3u << 22)      // pad_interval = 3 -> every 16 DWORDs
              | (3u << 25));    // pad_amount  = 3 -> 4 DWORDs
  g1[1] = (int)((td0 & 0xffffu) << 16);                 // tensor_dim0[15:0]
  g1[2] = (int)((td0 >> 16) | ((td1 & 0xffffu) << 16)); // td0 hi | td1 lo
  g1[3] = (int)((td1 >> 16) | (32u << 16));             // td1 hi | tile_dim0=32
  g1[4] = (int)(16u | ((unsigned)groups << 16));        // tile_dim1 | tile_dim2
  g1[5] = (int)rowStride;                               // dim0_stride[31:0]
  g1[6] = (int)((grpStride & 0xffffu) << 16);           // dim0 hi=0 | dim1 lo
  g1[7] = (int)(grpStride >> 16);                       // dim1_stride[47:16]
  i32x4 g2;
  g2[0] = 0x40000000;  // tensor_dim2
  g2[1] = 0x40000000;  // tensor_dim3 (tile_dim3 = 0 -> unused)
  g2[2] = 0;           // tensor_dim2_stride lo
  g2[3] = 0;           // tensor_dim2_stride hi | tile_dim3 = 0
  i32x4 g3;
  g3[0] = 0; g3[1] = 0; g3[2] = 0; g3[3] = 0;
  i32x8 g4;            // extra operand of the 6-arg (clang-23) builtin form;
#pragma unroll         // unused for this non-gather, non-iterate descriptor
  for (int i = 0; i < 8; ++i) g4[i] = 0;
  __builtin_amdgcn_tensor_load_to_lds(g0, g1, g2, g3, g4, 0);
}
#endif

// ---------------- conversion / init kernels ----------------

__global__ void cvt_f16_kernel(const float* __restrict__ src,
                               _Float16* __restrict__ dst, int n) {
  int i = blockIdx.x * blockDim.x + threadIdx.x;
  if (i < n) dst[i] = (_Float16)src[i];
}

// dst[n * dstStride + k] = (n < srcN && k < srcK) ? src[k * srcN + n] : 0
__global__ void transpose_cvt_kernel(const float* __restrict__ src,
                                     _Float16* __restrict__ dst,
                                     int srcK, int srcN, int dstStride, int total) {
  int i = blockIdx.x * blockDim.x + threadIdx.x;
  if (i >= total) return;
  int n = i / dstStride;
  int k = i - n * dstStride;
  float v = (n < srcN && k < srcK) ? src[(size_t)k * srcN + n] : 0.0f;
  dst[i] = (_Float16)v;
}

__global__ void init_state_kernel(float* __restrict__ c, _Float16* __restrict__ hF,
                                  _Float16* __restrict__ peF,
                                  const float* __restrict__ embed) {
  int i = blockIdx.x * blockDim.x + threadIdx.x;
  if (i < B_ROWS * H_DIM) { c[i] = 0.0f; hF[i] = (_Float16)0.0f; }
  if (i < B_ROWS * E_PAD) {
    int k = i % E_PAD;
    peF[i] = (k < E_DIM) ? (_Float16)embed[k] : (_Float16)0.0f;  // embed row 0
  }
}

// ---------------- big precompute GEMM: X = seq @ Wi[0:DIN] + bi ----------------
// Off the sequential chain; 2/3 of all FLOPs. Wave = 32x64 tile, block = 128x64.
__global__ void __launch_bounds__(128)
gemm_x_kernel(const _Float16* __restrict__ A, const _Float16* __restrict__ BT,
              const float* __restrict__ bias, float* __restrict__ X) {
  const int lane = threadIdx.x & 31;
  const int wave = threadIdx.x >> 5;
  const int n0 = blockIdx.x * 64;
  const int m0 = blockIdx.y * 128 + wave * 32;

  v8f acc[2][4];
#pragma unroll
  for (int i = 0; i < 2; ++i)
#pragma unroll
    for (int j = 0; j < 4; ++j) acc[i][j] = zero8();

  for (int k0 = 0; k0 < DIN; k0 += 32) {
    v16h a0 = load_frag16(A, DIN, m0, k0);
    v16h a1 = load_frag16(A, DIN, m0 + 16, k0);
#pragma unroll
    for (int j = 0; j < 4; ++j) {
      v16h b = load_frag16(BT, KTI, n0 + 16 * j, k0);
      acc[0][j] = wmma_f16(a0, b, acc[0][j]);
      acc[1][j] = wmma_f16(a1, b, acc[1][j]);
    }
  }

  const int mb = (lane >> 4) << 3;
#pragma unroll
  for (int j = 0; j < 4; ++j) {
    int n = n0 + 16 * j + (lane & 15);
    float bv = bias[n];
#pragma unroll
    for (int i = 0; i < 2; ++i)
#pragma unroll
      for (int r = 0; r < 8; ++r)
        X[(size_t)(m0 + 16 * i + mb + r) * N6H + n] = acc[i][j][r] + bv;
  }
}

// ---------------- fused per-step kernel ----------------
// Each block owns 16 H-columns across ALL SIX gate segments, so the LSTM +
// highway elementwise math fuses into the GEMM epilogue. Weight tiles are
// staged into a double-buffered LDS tile: on the TDM path, wave 0 issues an
// async tensor_load_to_lds for K-step k+1 while all 8 waves compute WMMAs
// from the current buffer, then s_wait_tensorcnt + one barrier. Fallback:
// cooperative vector staging. h is ping-pong buffered across timesteps.
__global__ void __launch_bounds__(256)
step_fused_kernel(const _Float16* __restrict__ hIn, _Float16* __restrict__ hOut,
                  const _Float16* __restrict__ peF,
                  const _Float16* __restrict__ WiT, const _Float16* __restrict__ WsT,
                  const float* __restrict__ Xt, const float* __restrict__ bs,
                  float* __restrict__ Cst) {
  __shared__ _Float16 ldsB[2][TILE_HALVES];  // 2 x 7680 B

  const int tid = threadIdx.x;
  const int lane = tid & 31;
  const int m0 = (tid >> 5) * 16;      // 8 waves cover M = 128
  const int col0 = blockIdx.x * 16;    // 16 H-columns per block (grid = 64)

  v8f acc[6];
#pragma unroll
  for (int g = 0; g < 6; ++g) acc[g] = zero8();

#if USE_TDM
  const unsigned ldsBase = (unsigned)(unsigned long long)&ldsB[0][0];
  const unsigned bufBytes = TILE_HALVES * (unsigned)sizeof(_Float16);
  const _Float16* peW = WiT + (size_t)col0 * KTI + DIN;   // pe-part tile origin
  const _Float16* hW  = WsT + (size_t)col0 * H_DIM;       // h-part tile origin

  // ---- feedback-embedding part: K = 224, all 6 gate segments ----
  if (tid < 32) {
    tdm_load_tile(ldsBase, peW, 6, KTI, 1024u * KTI);
    __builtin_amdgcn_s_wait_tensorcnt(0);
  }
  __syncthreads();
  for (int k0 = 0; k0 < E_PAD; k0 += 32) {
    const int cur = (k0 >> 5) & 1;
    const bool more = (k0 + 32) < E_PAD;
    if (more && tid < 32)
      tdm_load_tile(ldsBase + (cur ^ 1) * bufBytes, peW + k0 + 32,
                    6, KTI, 1024u * KTI);
    v16h a = load_frag16(peF, E_PAD, m0, k0);
#pragma unroll
    for (int g = 0; g < 6; ++g) {
      v16h b = lds_frag16(&ldsB[cur][g * 16 * LDS_PITCH]);
      acc[g] = wmma_f16(a, b, acc[g]);
    }
    if (more && tid < 32) __builtin_amdgcn_s_wait_tensorcnt(0);
    __syncthreads();
  }

  // ---- recurrent-state part: K = 1024, gate segments 0..4 only ----
  if (tid < 32) {
    tdm_load_tile(ldsBase, hW, 5, H_DIM, 1024u * H_DIM);
    __builtin_amdgcn_s_wait_tensorcnt(0);
  }
  __syncthreads();
  for (int k0 = 0; k0 < H_DIM; k0 += 32) {
    const int cur = (k0 >> 5) & 1;
    const bool more = (k0 + 32) < H_DIM;
    if (more && tid < 32)
      tdm_load_tile(ldsBase + (cur ^ 1) * bufBytes, hW + k0 + 32,
                    5, H_DIM, 1024u * H_DIM);
    v16h a = load_frag16(hIn, H_DIM, m0, k0);
#pragma unroll
    for (int g = 0; g < 5; ++g) {
      v16h b = lds_frag16(&ldsB[cur][g * 16 * LDS_PITCH]);
      acc[g] = wmma_f16(a, b, acc[g]);
    }
    if (more && tid < 32) __builtin_amdgcn_s_wait_tensorcnt(0);
    __syncthreads();
  }
#else
  // ---- fallback: cooperative vector staging (round-2 path) ----
  for (int k0 = 0; k0 < E_PAD; k0 += 32) {
    for (int i = tid; i < 96 * 4; i += 256) {
      int row = i >> 2, chunk = i & 3;
      int n = ((row >> 4) << 10) + col0 + (row & 15);
      const _Float16* src = WiT + (size_t)n * KTI + DIN + k0 + chunk * 8;
      *(v8h*)&ldsB[0][row * LDS_PITCH + chunk * 8] = *(const v8h*)src;
    }
    __syncthreads();
    v16h a = load_frag16(peF, E_PAD, m0, k0);
#pragma unroll
    for (int g = 0; g < 6; ++g)
      acc[g] = wmma_f16(a, lds_frag16(&ldsB[0][g * 16 * LDS_PITCH]), acc[g]);
    __syncthreads();
  }
  for (int k0 = 0; k0 < H_DIM; k0 += 32) {
    for (int i = tid; i < 80 * 4; i += 256) {
      int row = i >> 2, chunk = i & 3;
      int n = ((row >> 4) << 10) + col0 + (row & 15);
      const _Float16* src = WsT + (size_t)n * H_DIM + k0 + chunk * 8;
      *(v8h*)&ldsB[0][row * LDS_PITCH + chunk * 8] = *(const v8h*)src;
    }
    __syncthreads();
    v16h a = load_frag16(hIn, H_DIM, m0, k0);
#pragma unroll
    for (int g = 0; g < 5; ++g)
      acc[g] = wmma_f16(a, lds_frag16(&ldsB[0][g * 16 * LDS_PITCH]), acc[g]);
    __syncthreads();
  }
#endif

  // ---- fused LSTM + highway epilogue ----
  const int colL = lane & 15;
  const int mb = (lane >> 4) << 3;
  const int col = col0 + colL;
#pragma unroll
  for (int r = 0; r < 8; ++r) {
    int row = m0 + mb + r;
    size_t xb = (size_t)row * N6H + col;
    float g0 = acc[0][r] + Xt[xb]              + bs[col];
    float g1 = acc[1][r] + Xt[xb + 1 * H_DIM]  + bs[col + 1 * H_DIM];
    float g2 = acc[2][r] + Xt[xb + 2 * H_DIM]  + bs[col + 2 * H_DIM];
    float g3 = acc[3][r] + Xt[xb + 3 * H_DIM]  + bs[col + 3 * H_DIM];
    float g4 = acc[4][r] + Xt[xb + 4 * H_DIM]  + bs[col + 4 * H_DIM];
    float g5 = acc[5][r] + Xt[xb + 5 * H_DIM];           // highway: bi only
    float i_g = sigmoidf_(g0);
    float f_g = sigmoidf_(g1);
    float m_i = tanhf(g2);
    float o_g = sigmoidf_(g3);
    float hw  = sigmoidf_(g4);
    size_t hi = (size_t)row * H_DIM + col;
    float c_new = i_g * m_i + f_g * Cst[hi];
    float outv = o_g * tanhf(c_new);
    float h_new = hw * outv + (1.0f - hw) * g5;
    Cst[hi] = c_new;
    hOut[hi] = (_Float16)h_new;
  }
}

// ---------------- fused classifier + argmax + feedback kernel ----------------
__global__ void __launch_bounds__(256)
pred_final_kernel(const _Float16* __restrict__ hF, const _Float16* __restrict__ WoT,
                  const float* __restrict__ bo, const int* __restrict__ labels,
                  const float* __restrict__ embed, _Float16* __restrict__ peF,
                  float* __restrict__ dists, float* __restrict__ comms, int t) {
  __shared__ int l2e_s[B_ROWS];

  const int tid = threadIdx.x;
  const int lane = tid & 31;
  const int m0 = (tid >> 5) * 16;

  v8f acc[10];
#pragma unroll
  for (int j = 0; j < 10; ++j) acc[j] = zero8();

  for (int k0 = 0; k0 < H_DIM; k0 += 32) {
    v16h a = load_frag16(hF, H_DIM, m0, k0);
#pragma unroll
    for (int j = 0; j < 10; ++j) {
      v16h b = load_frag16(WoT, H_DIM, 16 * j, k0);
      acc[j] = wmma_f16(a, b, acc[j]);
    }
  }

  const int colL = lane & 15;
  const int mb = (lane >> 4) << 3;
#pragma unroll
  for (int r = 0; r < 8; ++r) {
    int row = m0 + mb + r;
    float best = -3.0e38f; int bidx = 1;
#pragma unroll
    for (int j = 0; j < 10; ++j) {
      int n = 16 * j + colL;
      if (n < C_CLS) {
        float v = acc[j][r] + bo[n];
        dists[((size_t)t * B_ROWS + row) * C_CLS + n] = v;
        if (n >= 1 && v > best) { best = v; bidx = n; }
      }
    }
#pragma unroll
    for (int msk = 1; msk < 16; msk <<= 1) {
      float ob = __shfl_xor(best, msk);
      int   oi = __shfl_xor(bidx, msk);
      if (ob > best || (ob == best && oi < bidx)) { best = ob; bidx = oi; }
    }
    if (colL == 0) {
      int lt = labels[t * B_ROWS + row];
      l2e_s[row] = (lt == 0) ? bidx : lt;
    }
  }
  __syncthreads();

  for (int i = tid; i < B_ROWS; i += 256)
    comms[t * B_ROWS + i] = (float)l2e_s[i];
  for (int i = tid; i < B_ROWS * E_PAD; i += 256) {
    int b = i / E_PAD, k = i - b * E_PAD;
    peF[i] = (k < E_DIM) ? (_Float16)embed[(size_t)(l2e_s[b] + 1) * E_DIM + k]
                         : (_Float16)0.0f;
  }
}

// ---------------- host launch ----------------

extern "C" void kernel_launch(void* const* d_in, const int* in_sizes, int n_in,
                              void* d_out, int out_size, void* d_ws, size_t ws_size,
                              hipStream_t stream) {
  (void)in_sizes; (void)n_in; (void)out_size; (void)ws_size;
  const float* seq    = (const float*)d_in[0];
  const int*   labels = (const int*)d_in[1];
  const float* embed  = (const float*)d_in[2];
  const float* Wi     = (const float*)d_in[3];
  const float* bi     = (const float*)d_in[4];
  const float* Ws     = (const float*)d_in[5];
  const float* bs     = (const float*)d_in[6];
  const float* Wo     = (const float*)d_in[7];
  const float* bo     = (const float*)d_in[8];

  float* dists = (float*)d_out;
  float* comms = dists + (size_t)T_STEPS * B_ROWS * C_CLS;

  char* ws = (char*)d_ws;
  size_t off = 0;
  auto alloc = [&](size_t bytes) -> char* {
    char* p = ws + off;
    off = (off + bytes + 255) & ~(size_t)255;
    return p;
  };
  _Float16* seqF  = (_Float16*)alloc((size_t)T_STEPS * B_ROWS * DIN * 2);
  _Float16* WiT   = (_Float16*)alloc((size_t)N6H * KTI * 2);
  _Float16* WsT   = (_Float16*)alloc((size_t)N5H * H_DIM * 2);
  _Float16* WoT   = (_Float16*)alloc((size_t)NWO * H_DIM * 2);
  float*    X     = (float*)   alloc((size_t)T_STEPS * B_ROWS * N6H * 4);
  float*    Cst   = (float*)   alloc((size_t)B_ROWS * H_DIM * 4);
  _Float16* hBuf0 = (_Float16*)alloc((size_t)B_ROWS * H_DIM * 2);
  _Float16* hBuf1 = (_Float16*)alloc((size_t)B_ROWS * H_DIM * 2);
  _Float16* peF   = (_Float16*)alloc((size_t)B_ROWS * E_PAD * 2);

  {
    int n = T_STEPS * B_ROWS * DIN;
    cvt_f16_kernel<<<(n + 255) / 256, 256, 0, stream>>>(seq, seqF, n);
  }
  {
    int total = N6H * KTI;   // WiT: [6144, 2272] from Wi [2248, 6144]
    transpose_cvt_kernel<<<(total + 255) / 256, 256, 0, stream>>>(
        Wi, WiT, DIN + E_DIM, N6H, KTI, total);
  }
  {
    int total = N5H * H_DIM; // WsT: [5120, 1024] from Ws [1024, 5120]
    transpose_cvt_kernel<<<(total + 255) / 256, 256, 0, stream>>>(
        Ws, WsT, H_DIM, N5H, H_DIM, total);
  }
  {
    int total = NWO * H_DIM; // WoT: [160, 1024] from Wo [1024, 151]
    transpose_cvt_kernel<<<(total + 255) / 256, 256, 0, stream>>>(
        Wo, WoT, H_DIM, C_CLS, H_DIM, total);
  }
  init_state_kernel<<<(B_ROWS * H_DIM + 255) / 256, 256, 0, stream>>>(
      Cst, hBuf0, peF, embed);

  {
    dim3 grid(N6H / 64, (T_STEPS * B_ROWS) / 128);
    gemm_x_kernel<<<grid, 128, 0, stream>>>(seqF, WiT, bi, X);
  }

  for (int t = 0; t < T_STEPS; ++t) {
    const float* Xt = X + (size_t)t * B_ROWS * N6H;
    _Float16* hIn  = (t & 1) ? hBuf1 : hBuf0;
    _Float16* hOut = (t & 1) ? hBuf0 : hBuf1;
    step_fused_kernel<<<H_DIM / 16, 256, 0, stream>>>(
        hIn, hOut, peF, WiT, WsT, Xt, bs, Cst);
    pred_final_kernel<<<1, 256, 0, stream>>>(
        hOut, WoT, bo, labels, embed, peF, dists, comms, t);
  }
}
